// CSPVNet_33629593927946
// MI455X (gfx1250) — matrix operand order repaired and checked by reading, not compiled, not run on hardware
//
#include <hip/hip_runtime.h>
#include <hip/hip_bf16.h>
#include <math.h>

#define HDIM 128
#define NFREQ 10
#define DIS 60
#define LAYERS 4
#define NN 50000
#define EE 800000
#define GG 1000
#define TDIM 128
#define PI_F 3.14159265358979f

typedef __attribute__((ext_vector_type(16))) __bf16 v16bf;
typedef __attribute__((ext_vector_type(8)))  __bf16 v8bf;
typedef __attribute__((ext_vector_type(8)))  float  v8f;

__device__ __forceinline__ float siluf(float x) { return x / (1.f + __expf(-x)); }

// ---- WMMA tile helpers (wave32, 16x16x32 bf16, f32 accum) -------------------
// A layout (16-bit A 16x32): lanes 0-15 -> M=lane, K in {0..7, 16..23};
// lanes 16-31 -> M=lane-16, K in {8..15, 24..31}.
__device__ __forceinline__ v16bf load_a_tile(const __bf16* A, int kpad, int lane, int kt) {
    int m = lane & 15;
    int koff = (lane < 16) ? 0 : 8;
    const __bf16* p = A + m * kpad + kt * 32 + koff;
    v8bf lo = *(const v8bf*)(p);
    v8bf hi = *(const v8bf*)(p + 16);
    return __builtin_shufflevector(lo, hi, 0,1,2,3,4,5,6,7,8,9,10,11,12,13,14,15);
}
// B pre-swizzled: tile t -> 32 lanes x 16 contiguous bf16 each.
__device__ __forceinline__ v16bf load_b_tile(const __bf16* B, int tile, int lane) {
    const __bf16* p = B + (size_t)tile * 512 + lane * 16;
    v8bf lo = *(const v8bf*)(p);
    v8bf hi = *(const v8bf*)(p + 8);
    return __builtin_shufflevector(lo, hi, 0,1,2,3,4,5,6,7,8,9,10,11,12,13,14,15);
}
// One wave computes two adjacent 16x16 N-tiles (nt0, nt0+1), bias-initialized.
__device__ __forceinline__ void gemm_pair(const __bf16* A, int kpad, const __bf16* B,
                                          int ntilesN, int nt0, int ktiles,
                                          const float* bias, int lane,
                                          v8f& acc0, v8f& acc1) {
    float b0 = bias[nt0 * 16 + (lane & 15)];
    float b1 = bias[nt0 * 16 + 16 + (lane & 15)];
#pragma unroll
    for (int r = 0; r < 8; ++r) { acc0[r] = b0; acc1[r] = b1; }
    for (int kt = 0; kt < ktiles; ++kt) {
        v16bf a  = load_a_tile(A, kpad, lane, kt);
        v16bf bA = load_b_tile(B, kt * ntilesN + nt0, lane);
        v16bf bB = load_b_tile(B, kt * ntilesN + nt0 + 1, lane);
        acc0 = __builtin_amdgcn_wmma_f32_16x16x32_bf16(false, a, false, bA, (short)0, acc0, false, false);
        acc1 = __builtin_amdgcn_wmma_f32_16x16x32_bf16(false, a, false, bB, (short)0, acc1, false, false);
    }
}
// 2x2 tile: two M-tiles (A, A+16*kpad) x two N-tiles; B reused across M-tiles.
__device__ __forceinline__ void gemm_2x2(const __bf16* A, int kpad, const __bf16* B,
                                         int ntilesN, int nt0, int ktiles,
                                         const float* bias, int lane,
                                         v8f& c00, v8f& c01, v8f& c10, v8f& c11) {
    float b0 = bias[nt0 * 16 + (lane & 15)];
    float b1 = bias[nt0 * 16 + 16 + (lane & 15)];
#pragma unroll
    for (int r = 0; r < 8; ++r) { c00[r] = b0; c01[r] = b1; c10[r] = b0; c11[r] = b1; }
    const __bf16* A1 = A + 16 * kpad;
    for (int kt = 0; kt < ktiles; ++kt) {
        v16bf a0  = load_a_tile(A,  kpad, lane, kt);
        v16bf a1  = load_a_tile(A1, kpad, lane, kt);
        v16bf bb0 = load_b_tile(B, kt * ntilesN + nt0, lane);
        v16bf bb1 = load_b_tile(B, kt * ntilesN + nt0 + 1, lane);
        c00 = __builtin_amdgcn_wmma_f32_16x16x32_bf16(false, a0, false, bb0, (short)0, c00, false, false);
        c01 = __builtin_amdgcn_wmma_f32_16x16x32_bf16(false, a0, false, bb1, (short)0, c01, false, false);
        c10 = __builtin_amdgcn_wmma_f32_16x16x32_bf16(false, a1, false, bb0, (short)0, c10, false, false);
        c11 = __builtin_amdgcn_wmma_f32_16x16x32_bf16(false, a1, false, bb1, (short)0, c11, false, false);
    }
}

// ---- weight swizzle: W [Nout rows][Kin] f32 -> B-operand bf16 tiles ---------
__global__ void k_swizzle(const float* __restrict__ W, __bf16* __restrict__ out,
                          int Kin, int Ntiles) {
    int tile = blockIdx.x;           // kt*Ntiles + nt
    int lane = threadIdx.x;          // 32
    int kt = tile / Ntiles, nt = tile % Ntiles;
    int n = nt * 16 + (lane & 15);
    int kbase = kt * 32 + ((lane < 16) ? 0 : 8);
    __bf16* o = out + (size_t)tile * 512 + lane * 16;
#pragma unroll
    for (int e = 0; e < 16; ++e) {
        int k = kbase + e + ((e >= 8) ? 8 : 0);
        float v = (k < Kin) ? W[(size_t)n * Kin + k] : 0.f;
        o[e] = (__bf16)v;
    }
}

__global__ void k_zero(float* p, size_t n) {
    size_t i = (size_t)blockIdx.x * blockDim.x + threadIdx.x;
    size_t stride = (size_t)gridDim.x * blockDim.x;
    for (; i < n; i += stride) p[i] = 0.f;
}

// ---- time embedding ---------------------------------------------------------
__global__ void k_temb(const float* __restrict__ t, const float* __restrict__ timeB,
                       __bf16* __restrict__ temb) {
    int g = blockIdx.x, j = threadIdx.x;   // 64 threads
    float p = 2.f * PI_F * t[g] * timeB[j];
    temb[(size_t)g * TDIM + j]      = (__bf16)__sinf(p);
    temb[(size_t)g * TDIM + 64 + j] = (__bf16)__cosf(p);
}

// ---- per-edge static features: l_edge(6) + pd(60) bf16, dv f32, node counts -
__global__ void k_edge_static(const float* __restrict__ pos, const float* __restrict__ vel,
                              const float* __restrict__ l, const int* __restrict__ node_index,
                              const int* __restrict__ src, const int* __restrict__ dst,
                              __bf16* __restrict__ estat, float* __restrict__ dvout,
                              float* __restrict__ cnt) {
    int e = blockIdx.x * 256 + threadIdx.x;
    if (e >= EE) return;
    int s = src[e], d = dst[e];
    float pd3[3], dv3[3];
#pragma unroll
    for (int c = 0; c < 3; ++c) {
        pd3[c] = pos[(size_t)d * 3 + c] - pos[(size_t)s * 3 + c];
        dv3[c] = vel[(size_t)d * 3 + c] - vel[(size_t)s * 3 + c];
        dvout[(size_t)e * 3 + c] = dv3[c];
    }
    int g = node_index[s];
    __bf16* ep = estat + (size_t)e * 66;
#pragma unroll
    for (int c = 0; c < 6; ++c) ep[c] = (__bf16)l[(size_t)g * 6 + c];
#pragma unroll
    for (int c = 0; c < 3; ++c) {
        float fr = PI_F;
        for (int k = 0; k < NFREQ; ++k) {
            float a = pd3[c] * fr;
            ep[6 + c * 2 * NFREQ + k]         = (__bf16)__sinf(a);
            ep[6 + c * 2 * NFREQ + NFREQ + k] = (__bf16)__cosf(a);
            fr *= 2.f;
        }
    }
    atomicAdd(&cnt[s], 1.f);
}

// ---- atom embedding MLP: nf = [emb(h), temb(g)] @ atom_w.T + b  (WMMA) ------
__global__ __launch_bounds__(128) void k_node_init(
        const float* __restrict__ emb, const int* __restrict__ h,
        const int* __restrict__ node_index, const __bf16* __restrict__ temb,
        const __bf16* __restrict__ wsw, const float* __restrict__ bias,
        float* __restrict__ nf) {
    __shared__ __attribute__((aligned(32))) __bf16 A[16 * 256];
    int tid = threadIdx.x, lane = tid & 31, wv = tid >> 5;
    int n0 = blockIdx.x * 16;
    for (int idx = tid; idx < 16 * 256; idx += 128) {
        int m = idx >> 8, f = idx & 255;
        int node = n0 + m;
        float val = (f < HDIM) ? emb[(size_t)h[node] * HDIM + f]
                               : (float)temb[(size_t)node_index[node] * TDIM + (f - HDIM)];
        A[idx] = (__bf16)val;
    }
    __syncthreads();
    int nt0 = wv * 2;
    v8f a0, a1;
    gemm_pair(A, 256, wsw, 8, nt0, 8, bias, lane, a0, a1);
    int mb = (lane < 16) ? 0 : 8, nn = lane & 15;
#pragma unroll
    for (int r = 0; r < 8; ++r) {
        size_t i0 = (size_t)(n0 + mb + r) * HDIM;
        nf[i0 + nt0 * 16 + nn]       = a0[r];
        nf[i0 + (nt0 + 1) * 16 + nn] = a1[r];
    }
}

// ---- layernorm: wave32 shuffle reduce, 8 nodes / 256-thread block -----------
__global__ __launch_bounds__(256) void k_layernorm(
        const float* __restrict__ x, const float* __restrict__ g,
        const float* __restrict__ b, __bf16* __restrict__ y16,
        float* __restrict__ y32, int n) {
    int wv = threadIdx.x >> 5, lane = threadIdx.x & 31;
    int node = blockIdx.x * 8 + wv;
    if (node >= n) return;
    const float* xp = x + (size_t)node * HDIM;
    float4 v4 = *(const float4*)(xp + lane * 4);
    float s = v4.x + v4.y + v4.z + v4.w;
#pragma unroll
    for (int off = 16; off; off >>= 1) s += __shfl_xor(s, off);
    float mean = s * (1.f / 128.f);
    float d0 = v4.x - mean, d1 = v4.y - mean, d2 = v4.z - mean, d3 = v4.w - mean;
    float vs = d0 * d0 + d1 * d1 + d2 * d2 + d3 * d3;
#pragma unroll
    for (int off = 16; off; off >>= 1) vs += __shfl_xor(vs, off);
    float inv = rsqrtf(vs * (1.f / 128.f) + 1e-5f);
    float dd[4] = {d0, d1, d2, d3};
#pragma unroll
    for (int j = 0; j < 4; ++j) {
        int f = lane * 4 + j;
        float o = dd[j] * inv * g[f] + b[f];
        y16[(size_t)node * HDIM + f] = (__bf16)o;
        if (y32) y32[(size_t)node * HDIM + f] = o;
    }
}

// ---- edge MLP layer: 32 edges/block, 2x2 WMMA tiling, atomic scatter --------
__global__ __launch_bounds__(128) void k_edge_layer(
        const __bf16* __restrict__ hn16, const __bf16* __restrict__ estat,
        const float* __restrict__ dv, const int* __restrict__ src,
        const int* __restrict__ dst, const float* __restrict__ vprojW,
        const float* __restrict__ vprojB, const __bf16* __restrict__ w1,
        const float* __restrict__ b1, const __bf16* __restrict__ w2,
        const float* __restrict__ b2, float* __restrict__ aggsum) {
    __shared__ __attribute__((aligned(32))) __bf16 ein[32 * 384];   // 24 KB
    __shared__ __attribute__((aligned(32))) __bf16 mid[32 * 128];   // 8 KB
    __shared__ int ssrc[32], sdst[32];
    int tid = threadIdx.x, lane = tid & 31, wv = tid >> 5;
    int e0 = blockIdx.x * 32;
    // warm both weight stages while we stage activations (global_prefetch_b8)
    __builtin_prefetch((const char*)w1 + tid * 768, 0, 3);
    __builtin_prefetch((const char*)w2 + tid * 256, 0, 3);
    if (tid < 32) { ssrc[tid] = src[e0 + tid]; sdst[tid] = dst[e0 + tid]; }
    __syncthreads();
    // 4 threads per edge-row: vectorized hi/hj copies + scalar tail
    {
        int m = tid >> 2, q = tid & 3;
        int e = e0 + m;
        const uint4* hs = (const uint4*)(hn16 + (size_t)ssrc[m] * HDIM);
        const uint4* hd = (const uint4*)(hn16 + (size_t)sdst[m] * HDIM);
        uint4* ei = (uint4*)(ein + m * 384);
        uint4* ej = (uint4*)(ein + m * 384 + 128);
#pragma unroll
        for (int j = 0; j < 4; ++j) {
            ei[q * 4 + j] = hs[q * 4 + j];
            ej[q * 4 + j] = hd[q * 4 + j];
        }
        float dv0 = dv[(size_t)e * 3], dv1 = dv[(size_t)e * 3 + 1], dv2 = dv[(size_t)e * 3 + 2];
        for (int j = 0; j < 32; ++j) {
            int f = 256 + q * 32 + j;
            float val;
            if (f < 262) {
                val = (float)estat[(size_t)e * 66 + (f - 256)];
            } else if (f < 322) {
                int jj = f - 262;
                val = dv0 * vprojW[jj * 3] + dv1 * vprojW[jj * 3 + 1] +
                      dv2 * vprojW[jj * 3 + 2] + vprojB[jj];
            } else if (f < 382) {
                val = (float)estat[(size_t)e * 66 + 6 + (f - 322)];
            } else {
                val = 0.f;
            }
            ein[m * 384 + f] = (__bf16)val;
        }
    }
    __syncthreads();
    int nt0 = wv * 2;
    v8f c00, c01, c10, c11;
    gemm_2x2(ein, 384, w1, 8, nt0, 12, b1, lane, c00, c01, c10, c11);
    int mb = (lane < 16) ? 0 : 8, nn = lane & 15;
#pragma unroll
    for (int r = 0; r < 8; ++r) {
        mid[(mb + r) * 128 + nt0 * 16 + nn]            = (__bf16)siluf(c00[r]);
        mid[(mb + r) * 128 + (nt0 + 1) * 16 + nn]      = (__bf16)siluf(c01[r]);
        mid[(16 + mb + r) * 128 + nt0 * 16 + nn]       = (__bf16)siluf(c10[r]);
        mid[(16 + mb + r) * 128 + (nt0 + 1) * 16 + nn] = (__bf16)siluf(c11[r]);
    }
    __syncthreads();
    gemm_2x2(mid, 128, w2, 8, nt0, 4, b2, lane, c00, c01, c10, c11);
#pragma unroll
    for (int r = 0; r < 8; ++r) {
        int s0 = ssrc[mb + r], s1 = ssrc[16 + mb + r];
        atomicAdd(&aggsum[(size_t)s0 * HDIM + nt0 * 16 + nn],       siluf(c00[r]));
        atomicAdd(&aggsum[(size_t)s0 * HDIM + (nt0 + 1) * 16 + nn], siluf(c01[r]));
        atomicAdd(&aggsum[(size_t)s1 * HDIM + nt0 * 16 + nn],       siluf(c10[r]));
        atomicAdd(&aggsum[(size_t)s1 * HDIM + (nt0 + 1) * 16 + nn], siluf(c11[r]));
    }
}

// ---- node MLP layer: nin=[hn, agg_mean] -> 2x WMMA MLP -> residual ----------
__global__ __launch_bounds__(128) void k_node_layer(
        const __bf16* __restrict__ hn16, const float* __restrict__ aggsum,
        const float* __restrict__ cnt, const __bf16* __restrict__ w1,
        const float* __restrict__ b1, const __bf16* __restrict__ w2,
        const float* __restrict__ b2, float* __restrict__ nf) {
    __shared__ __attribute__((aligned(32))) __bf16 A[16 * 256];
    __shared__ __attribute__((aligned(32))) __bf16 mid[16 * 128];
    int tid = threadIdx.x, lane = tid & 31, wv = tid >> 5;
    int n0 = blockIdx.x * 16;
    for (int idx = tid; idx < 16 * 256; idx += 128) {
        int m = idx >> 8, f = idx & 255;
        size_t node = (size_t)(n0 + m);
        float val;
        if (f < 128) {
            val = (float)hn16[node * HDIM + f];
        } else {
            float c = cnt[node]; c = c < 1.f ? 1.f : c;
            val = aggsum[node * HDIM + (f - 128)] / c;
        }
        A[idx] = (__bf16)val;
    }
    __syncthreads();
    int nt0 = wv * 2;
    v8f a0, a1;
    gemm_pair(A, 256, w1, 8, nt0, 8, b1, lane, a0, a1);
    int mb = (lane < 16) ? 0 : 8, nn = lane & 15;
#pragma unroll
    for (int r = 0; r < 8; ++r) {
        mid[(mb + r) * 128 + nt0 * 16 + nn]       = (__bf16)siluf(a0[r]);
        mid[(mb + r) * 128 + (nt0 + 1) * 16 + nn] = (__bf16)siluf(a1[r]);
    }
    __syncthreads();
    gemm_pair(mid, 128, w2, 8, nt0, 4, b2, lane, a0, a1);
#pragma unroll
    for (int r = 0; r < 8; ++r) {
        size_t i0 = (size_t)(n0 + mb + r) * HDIM;
        nf[i0 + nt0 * 16 + nn]       += siluf(a0[r]);
        nf[i0 + (nt0 + 1) * 16 + nn] += siluf(a1[r]);
    }
}

// ---- graph feature sums (for ol head) ---------------------------------------
__global__ void k_gsum(const float* __restrict__ hnF32, const int* __restrict__ node_index,
                       float* __restrict__ gsum, float* __restrict__ cntg) {
    int i = blockIdx.x;      // node
    int f = threadIdx.x;     // 128
    int g = node_index[i];
    atomicAdd(&gsum[(size_t)g * HDIM + f], hnF32[(size_t)i * HDIM + f]);
    if (f == 0) atomicAdd(&cntg[g], 1.f);
}

// ---- outv head: silu(hn @ w1.T + b1) @ w2.T, plus per-graph sums ------------
__global__ __launch_bounds__(128) void k_outv(
        const __bf16* __restrict__ hn16, const __bf16* __restrict__ w1,
        const float* __restrict__ b1, const float* __restrict__ w2,
        const int* __restrict__ node_index, float* __restrict__ ovraw,
        float* __restrict__ ovsum) {
    __shared__ __attribute__((aligned(32))) __bf16 A[16 * 128];
    __shared__ float O[16 * 128];
    int tid = threadIdx.x, lane = tid & 31, wv = tid >> 5;
    int n0 = blockIdx.x * 16;
    for (int idx = tid; idx < 16 * 128; idx += 128) A[idx] = hn16[(size_t)n0 * HDIM + idx];
    __syncthreads();
    int nt0 = wv * 2;
    v8f a0, a1;
    gemm_pair(A, 128, w1, 8, nt0, 4, b1, lane, a0, a1);
    int mb = (lane < 16) ? 0 : 8, nn = lane & 15;
#pragma unroll
    for (int r = 0; r < 8; ++r) {
        O[(mb + r) * 128 + nt0 * 16 + nn]       = siluf(a0[r]);
        O[(mb + r) * 128 + (nt0 + 1) * 16 + nn] = siluf(a1[r]);
    }
    __syncthreads();
    if (tid < 48) {
        int m = tid / 3, c = tid % 3;
        float s = 0.f;
        for (int k = 0; k < 128; ++k) s += O[m * 128 + k] * w2[c * 128 + k];
        int node = n0 + m;
        ovraw[(size_t)node * 3 + c] = s;
        atomicAdd(&ovsum[(size_t)node_index[node] * 3 + c], s);
    }
}

__global__ void k_finalv(const float* __restrict__ ovraw, const float* __restrict__ ovsum,
                         const float* __restrict__ cntg, const int* __restrict__ node_index,
                         float* __restrict__ out) {
    int i = blockIdx.x * 256 + threadIdx.x;
    if (i >= NN) return;
    int g = node_index[i];
    float c = cntg[g]; c = c < 1.f ? 1.f : c;
#pragma unroll
    for (int j = 0; j < 3; ++j)
        out[(size_t)i * 3 + j] = ovraw[(size_t)i * 3 + j] - ovsum[(size_t)g * 3 + j] / c;
}

__global__ void k_outl(const float* __restrict__ gsum, const float* __restrict__ cntg,
                       const float* __restrict__ wl, float* __restrict__ out) {
    int idx = blockIdx.x * 64 + threadIdx.x;
    if (idx >= GG * 6) return;
    int g = idx / 6, c = idx % 6;
    float cn = cntg[g]; cn = cn < 1.f ? 1.f : cn;
    float s = 0.f;
    for (int k = 0; k < HDIM; ++k) s += (gsum[(size_t)g * HDIM + k] / cn) * wl[c * HDIM + k];
    out[(size_t)g * 6 + c] = s;
}

// -----------------------------------------------------------------------------
extern "C" void kernel_launch(void* const* d_in, const int* in_sizes, int n_in,
                              void* d_out, int out_size, void* d_ws, size_t ws_size,
                              hipStream_t stream) {
    (void)in_sizes; (void)n_in; (void)out_size; (void)ws_size;
    const float* t          = (const float*)d_in[0];
    const float* pos        = (const float*)d_in[1];
    const float* vel        = (const float*)d_in[2];
    const float* lmat       = (const float*)d_in[3];
    const int*   h          = (const int*)d_in[4];
    const int*   node_index = (const int*)d_in[5];
    const int*   eni        = (const int*)d_in[6];
    const float* time_B     = (const float*)d_in[7];
    const float* emb        = (const float*)d_in[8];
    const float* atom_w     = (const float*)d_in[9];
    const float* atom_b     = (const float*)d_in[10];
    const float* vproj_w    = (const float*)d_in[11];
    const float* vproj_b    = (const float*)d_in[12];
    const float* edge_w1    = (const float*)d_in[13];
    const float* edge_b1    = (const float*)d_in[14];
    const float* edge_w2    = (const float*)d_in[15];
    const float* edge_b2    = (const float*)d_in[16];
    const float* node_w1    = (const float*)d_in[17];
    const float* node_b1    = (const float*)d_in[18];
    const float* node_w2    = (const float*)d_in[19];
    const float* node_b2    = (const float*)d_in[20];
    const float* ln_g       = (const float*)d_in[21];
    const float* ln_b       = (const float*)d_in[22];
    const float* fln_g      = (const float*)d_in[23];
    const float* fln_b      = (const float*)d_in[24];
    const float* outv_w1    = (const float*)d_in[25];
    const float* outv_b1    = (const float*)d_in[26];
    const float* outv_w2    = (const float*)d_in[27];
    const float* outl_w     = (const float*)d_in[28];

    const int* src = eni;
    const int* dst = eni + EE;

    // workspace carve-out, 256B-aligned
    char* base = (char*)d_ws;
    size_t off = 0;
    auto carve = [&](size_t bytes) -> void* {
        void* p = base + off;
        off = (off + bytes + 255) & ~(size_t)255;
        return p;
    };
    float*  nf      = (float*)carve((size_t)NN * HDIM * 4);
    __bf16* hn16    = (__bf16*)carve((size_t)NN * HDIM * 2);
    float*  hnF32   = (float*)carve((size_t)NN * HDIM * 4);
    float*  aggsum  = (float*)carve((size_t)NN * HDIM * 4);
    float*  cnt     = (float*)carve((size_t)NN * 4);
    float*  cntg    = (float*)carve((size_t)GG * 4);
    float*  gsum    = (float*)carve((size_t)GG * HDIM * 4);
    float*  ovsum   = (float*)carve((size_t)GG * 3 * 4);
    float*  ovraw   = (float*)carve((size_t)NN * 3 * 4);
    __bf16* estat   = (__bf16*)carve((size_t)EE * 66 * 2);
    float*  dv      = (float*)carve((size_t)EE * 3 * 4);
    __bf16* temb    = (__bf16*)carve((size_t)GG * TDIM * 2);
    __bf16* atom_sw = (__bf16*)carve((size_t)8 * 8 * 512 * 2);
    __bf16* ew1_sw[LAYERS], *ew2_sw[LAYERS], *nw1_sw[LAYERS], *nw2_sw[LAYERS];
    for (int i = 0; i < LAYERS; ++i) {
        ew1_sw[i] = (__bf16*)carve((size_t)12 * 8 * 512 * 2);
        ew2_sw[i] = (__bf16*)carve((size_t)4 * 8 * 512 * 2);
        nw1_sw[i] = (__bf16*)carve((size_t)8 * 8 * 512 * 2);
        nw2_sw[i] = (__bf16*)carve((size_t)4 * 8 * 512 * 2);
    }
    __bf16* ov1_sw = (__bf16*)carve((size_t)4 * 8 * 512 * 2);

    // 1. swizzle weights into WMMA B-operand layout (bf16)
    k_swizzle<<<8 * 8, 32, 0, stream>>>(atom_w, atom_sw, 256, 8);
    for (int i = 0; i < LAYERS; ++i) {
        k_swizzle<<<12 * 8, 32, 0, stream>>>(edge_w1 + (size_t)i * 128 * 382, ew1_sw[i], 382, 8);
        k_swizzle<<<4 * 8, 32, 0, stream>>>(edge_w2 + (size_t)i * 128 * 128, ew2_sw[i], 128, 8);
        k_swizzle<<<8 * 8, 32, 0, stream>>>(node_w1 + (size_t)i * 128 * 256, nw1_sw[i], 256, 8);
        k_swizzle<<<4 * 8, 32, 0, stream>>>(node_w2 + (size_t)i * 128 * 128, nw2_sw[i], 128, 8);
    }
    k_swizzle<<<4 * 8, 32, 0, stream>>>(outv_w1, ov1_sw, 128, 8);

    // 2. static precompute
    k_zero<<<256, 256, 0, stream>>>(cnt, (size_t)NN);
    k_zero<<<64, 256, 0, stream>>>(cntg, (size_t)GG);
    k_zero<<<256, 256, 0, stream>>>(gsum, (size_t)GG * HDIM);
    k_zero<<<64, 256, 0, stream>>>(ovsum, (size_t)GG * 3);
    k_temb<<<GG, 64, 0, stream>>>(t, time_B, temb);
    k_edge_static<<<(EE + 255) / 256, 256, 0, stream>>>(pos, vel, lmat, node_index,
                                                        src, dst, estat, dv, cnt);
    // 3. atom embedding MLP (WMMA)
    k_node_init<<<NN / 16, 128, 0, stream>>>(emb, h, node_index, temb, atom_sw, atom_b, nf);

    // 4. message-passing layers
    for (int i = 0; i < LAYERS; ++i) {
        k_layernorm<<<(NN + 7) / 8, 256, 0, stream>>>(nf, ln_g + i * HDIM, ln_b + i * HDIM,
                                                      hn16, nullptr, NN);
        k_zero<<<1024, 256, 0, stream>>>(aggsum, (size_t)NN * HDIM);
        k_edge_layer<<<EE / 32, 128, 0, stream>>>(hn16, estat, dv, src, dst,
                                                  vproj_w + (size_t)i * DIS * 3,
                                                  vproj_b + (size_t)i * DIS,
                                                  ew1_sw[i], edge_b1 + i * HDIM,
                                                  ew2_sw[i], edge_b2 + i * HDIM, aggsum);
        k_node_layer<<<NN / 16, 128, 0, stream>>>(hn16, aggsum, cnt,
                                                  nw1_sw[i], node_b1 + i * HDIM,
                                                  nw2_sw[i], node_b2 + i * HDIM, nf);
    }

    // 5. final layernorm + heads
    k_layernorm<<<(NN + 7) / 8, 256, 0, stream>>>(nf, fln_g, fln_b, hn16, hnF32, NN);
    k_gsum<<<NN, 128, 0, stream>>>(hnF32, node_index, gsum, cntg);
    k_outv<<<NN / 16, 128, 0, stream>>>(hn16, ov1_sw, outv_b1, outv_w2, node_index,
                                        ovraw, ovsum);
    float* out_f = (float*)d_out;
    k_finalv<<<(NN + 255) / 256, 256, 0, stream>>>(ovraw, ovsum, cntg, node_index, out_f);
    k_outl<<<(GG * 6 + 63) / 64, 64, 0, stream>>>(gsum, cntg, outl_w, out_f + (size_t)NN * 3);
}
